// MultiHeadAttention_31808527794307
// MI455X (gfx1250) — compile-verified, matrix operand order checked by reading, not run on hardware
//
#include <hip/hip_runtime.h>
#include <hip/hip_bf16.h>

// MHA for MI455X (gfx1250): bf16 WMMA everywhere, f32 accumulation.
// B=2, T=2048, C=1024, H=16, D=64.

typedef __bf16 bf16_t;
typedef __attribute__((ext_vector_type(16))) __bf16 bf16x16;
typedef __attribute__((ext_vector_type(8)))  __bf16 bf16x8;
typedef __attribute__((ext_vector_type(8)))  float  v8f;
typedef unsigned int v4u  __attribute__((ext_vector_type(4)));
typedef int          v8i_ __attribute__((ext_vector_type(8)));
typedef int          v4i_ __attribute__((ext_vector_type(4)));

#define BB 2
#define TT 2048
#define CC 1024
#define HH 16
#define DD 64

#ifndef __has_builtin
#define __has_builtin(x) 0
#endif
#if __has_builtin(__builtin_amdgcn_tensor_load_to_lds)
#define HAVE_TDM 1
#else
#define HAVE_TDM 0
#endif

// Load one 16x32 bf16 A/B fragment per the CDNA5 16-bit layout:
// lane = (r = lane&15, hf = lane>>4); elements 0..7  <- K = hf*8 + 0..7
//                                     elements 8..15 <- K = 16 + hf*8 + 0..7
__device__ inline bf16x16 frag_ld(const bf16_t* rowk, int hf) {
  bf16x8 lo = *(const bf16x8*)(rowk + hf * 8);
  bf16x8 hi = *(const bf16x8*)(rowk + 16 + hf * 8);
  bf16x16 f;
#pragma unroll
  for (int i = 0; i < 8; ++i) { f[i] = lo[i]; f[8 + i] = hi[i]; }
  return f;
}

__device__ inline v8f wmma_bf16(bf16x16 a, bf16x16 b, v8f c) {
  return __builtin_amdgcn_wmma_f32_16x16x32_bf16(false, a, false, b, (short)0, c,
                                                 false, false);
}

// ---------------------------------------------------------------------------
// Kernel 1: fused Q/K/V projection.  Y = X @ W^T + b, written [B,H,T,D] bf16.
// blockIdx.z selects projection (0=Q with 0.125 scale, 1=K, 2=V).
// 128x64 block tile, K-step 32.  8 waves: wm = wave>>1 (32 rows), wn = wave&1
// (32 cols); each wave holds 4 accumulators (32x32), fragments reused 2x.
// ---------------------------------------------------------------------------
__global__ __launch_bounds__(256) void mha_proj_kernel(
    const float* __restrict__ q, const float* __restrict__ k,
    const float* __restrict__ v, const float* __restrict__ Wq,
    const float* __restrict__ bq, const float* __restrict__ Wk,
    const float* __restrict__ bk, const float* __restrict__ Wv,
    const float* __restrict__ bv, bf16_t* __restrict__ Qh,
    bf16_t* __restrict__ Kh, bf16_t* __restrict__ Vh) {
  const int p = blockIdx.z;
  const float* X    = (p == 0) ? q  : (p == 1) ? k  : v;
  const float* W    = (p == 0) ? Wq : (p == 1) ? Wk : Wv;
  const float* bias = (p == 0) ? bq : (p == 1) ? bk : bv;
  bf16_t* out       = (p == 0) ? Qh : (p == 1) ? Kh : Vh;
  const float scale = (p == 0) ? 0.125f : 1.0f;  // 1/sqrt(D) folded into Q

  __shared__ bf16_t Asm[128 * 40];  // X tile (row stride 40 = 80B, 16B-aligned)
  __shared__ bf16_t Bsm[64 * 40];   // W tile

  const int m0 = blockIdx.x * 128;  // over M = B*T = 4096
  const int n0 = blockIdx.y * 64;   // over N = C = 1024
  const int tid = threadIdx.x;
  const int wave = tid >> 5, lane = tid & 31;
  const int wm = wave >> 1, wn = wave & 1;
  const int hf = lane >> 4, ln = lane & 15;

  v8f acc[2][2];
#pragma unroll
  for (int a = 0; a < 2; ++a)
#pragma unroll
    for (int b = 0; b < 2; ++b)
#pragma unroll
      for (int i = 0; i < 8; ++i) acc[a][b][i] = 0.0f;

  for (int kt = 0; kt < CC; kt += 32) {
    __syncthreads();
#pragma unroll
    for (int it = 0; it < 4; ++it) {  // A: 128x32 f32 = 1024 float4
      int idx = tid + it * 256;
      int r = idx >> 3, c4 = (idx & 7) * 4;
      const float4 xa = *(const float4*)(X + (size_t)(m0 + r) * CC + kt + c4);
      bf16_t* da = &Asm[r * 40 + c4];
      da[0] = (bf16_t)xa.x; da[1] = (bf16_t)xa.y;
      da[2] = (bf16_t)xa.z; da[3] = (bf16_t)xa.w;
    }
#pragma unroll
    for (int it = 0; it < 2; ++it) {  // B: 64x32 f32 = 512 float4
      int idx = tid + it * 256;
      int r = idx >> 3, c4 = (idx & 7) * 4;
      const float4 wb = *(const float4*)(W + (size_t)(n0 + r) * CC + kt + c4);
      bf16_t* db = &Bsm[r * 40 + c4];
      db[0] = (bf16_t)wb.x; db[1] = (bf16_t)wb.y;
      db[2] = (bf16_t)wb.z; db[3] = (bf16_t)wb.w;
    }
    if (kt + 32 < CC) {  // prefetch next weight K-slab (global_prefetch_b8)
      __builtin_prefetch(W + (size_t)(n0 + (tid >> 2)) * CC + kt + 32, 0, 1);
    }
    __syncthreads();
    bf16x16 af[2], bfr[2];
#pragma unroll
    for (int a = 0; a < 2; ++a)
      af[a] = frag_ld(&Asm[(wm * 32 + a * 16 + ln) * 40], hf);
#pragma unroll
    for (int b = 0; b < 2; ++b)
      bfr[b] = frag_ld(&Bsm[(wn * 32 + b * 16 + ln) * 40], hf);
#pragma unroll
    for (int a = 0; a < 2; ++a)
#pragma unroll
      for (int b = 0; b < 2; ++b) acc[a][b] = wmma_bf16(af[a], bfr[b], acc[a][b]);
  }

  // Epilogue: C layout lane holds (m = i + 8*hf, n = ln); write [B,H,T,D].
#pragma unroll
  for (int a = 0; a < 2; ++a)
#pragma unroll
    for (int b = 0; b < 2; ++b) {
      const int n = n0 + wn * 32 + b * 16 + ln;
      const int h = n >> 6, d = n & 63;
      const float bn = bias[n];
#pragma unroll
      for (int i = 0; i < 8; ++i) {
        const int m = m0 + wm * 32 + a * 16 + i + 8 * hf;
        const int bb = m >> 11, t = m & 2047;
        out[(((size_t)bb * HH + h) * TT + t) * DD + d] =
            (bf16_t)((acc[a][b][i] + bn) * scale);
      }
    }
}

// ---------------------------------------------------------------------------
// Kernel 2: causal flash attention.  Each wave owns 16 query rows; block
// stages 64-row K tiles (TDM async DMA, HW-padded to 72-elem rows) and V
// tiles (transposed, manual) in LDS.  Computes S^T = K @ Q^T so the f32
// C-layout of the scores converts in-register to the bf16 B-fragment of P^T
// for O^T += V^T @ P^T.  Output written [B,T,H,D] bf16.
// ---------------------------------------------------------------------------
__global__ __launch_bounds__(256) void mha_attn_kernel(
    const bf16_t* __restrict__ Qh, const bf16_t* __restrict__ Kh,
    const bf16_t* __restrict__ Vh, bf16_t* __restrict__ AO) {
  const int bh = blockIdx.y;          // 0..31 = b*H + h
  const int qblk = blockIdx.x;        // 0..15, 128 query rows per block
  const bf16_t* Qb = Qh + (size_t)bh * TT * DD;
  const bf16_t* Kb = Kh + (size_t)bh * TT * DD;
  const bf16_t* Vb = Vh + (size_t)bh * TT * DD;
  const int b = bh >> 4, h = bh & 15;

  __shared__ bf16_t Ks[64 * 72];      // K tile [s][d], stride 72 (144B)
  __shared__ bf16_t Vt[64 * 72];      // V tile transposed [d][s]

  const int tid = threadIdx.x, wave = tid >> 5, lane = tid & 31;
  const int hf = lane >> 4, ln = lane & 15;
  const int qrow0 = qblk * 128 + wave * 16;
  const int qg = qrow0 + ln;          // this lane's query column (in S^T)

  // Q fragments for this wave (A-layout == B-layout for our loader).
  const bf16_t* qrow = Qb + (size_t)(qrow0 + ln) * DD;
  const bf16x16 qf0 = frag_ld(qrow, hf);       // d = 0..31
  const bf16x16 qf1 = frag_ld(qrow + 32, hf);  // d = 32..63

  v8f oacc[4];
#pragma unroll
  for (int j = 0; j < 4; ++j)
#pragma unroll
    for (int i = 0; i < 8; ++i) oacc[j][i] = 0.0f;
  float mx = -1e30f, denom = 0.0f;

#if HAVE_TDM
  // TDM descriptor pieces that are loop-invariant.
  // Group1: data_size=2B, pad_enable, pad_interval=32 DWORDs (=128B row),
  // pad_amount=4 DWORDs (=16B) -> LDS row stride 144B = 72 bf16.  Tensor is
  // [dim1=T rows, dim0=64], contiguous stride 64; tile = 64x64.
  const unsigned g1w0 =
      (1u << 16) | (1u << 20) | (4u << 22) | (3u << 25);  // 0x07110000
  const unsigned ldsK = (unsigned)(uintptr_t)&Ks[0];  // low 32 bits = LDS addr
#endif

  const int nst = (qblk + 1) * 2;  // 64-wide key tiles needed (causal)
  for (int st = 0; st < nst; ++st) {
    const int s0 = st * 64;
    __syncthreads();
#if HAVE_TDM
    if (wave == 0) {
      const unsigned long long ga =
          (unsigned long long)(uintptr_t)(Kb + (size_t)s0 * DD);
      v4u g0 = {1u /*count=1*/, ldsK, (unsigned)ga,
                (unsigned)((ga >> 32) & 0x1FFFFFFu) | 0x80000000u /*type=2*/};
      v8i_ g1 = {(int)g1w0,
                 (int)(64u << 16) /*tensor_dim0 lo<<16*/,
                 (int)(2048u << 16) /*tensor_dim1 lo<<16*/,
                 (int)(64u << 16) /*tile_dim0*/,
                 64 /*tile_dim1*/,
                 64 /*tensor_dim0_stride*/,
                 0, 0};
      v4i_ gz = {0, 0, 0, 0};
      v8i_ gz8 = {0, 0, 0, 0, 0, 0, 0, 0};
      // 6-arg variant (clang-23 / therock-10.0 headers).
      __builtin_amdgcn_tensor_load_to_lds(g0, g1, gz, gz, gz8, 0);
    }
#endif
#pragma unroll
    for (int it = 0; it < 2; ++it) {  // V tile: 512 chunks of 8 bf16
      int idx = tid + it * 256;
      int r = idx >> 3, c8 = (idx & 7) * 8;
#if !HAVE_TDM
      bf16x8 kk = *(const bf16x8*)(Kb + (size_t)(s0 + r) * DD + c8);
      *(bf16x8*)&Ks[r * 72 + c8] = kk;
#endif
      bf16x8 vv = *(const bf16x8*)(Vb + (size_t)(s0 + r) * DD + c8);
#pragma unroll
      for (int e = 0; e < 8; ++e) Vt[(c8 + e) * 72 + r] = vv[e];
    }
#if HAVE_TDM
    if (wave == 0) __builtin_amdgcn_s_wait_tensorcnt((short)0);
#endif
    __syncthreads();

    // Causal early-out: tile entirely above this wave's 16 query rows
    // (wave-uniform branch; EXEC stays all-ones around the WMMAs).
    if (s0 > qrow0 + 15) continue;

    // S^T tiles: rows m = s_local, cols n = q.  4 accs cover s0..s0+63.
    v8f sacc[4];
#pragma unroll
    for (int j = 0; j < 4; ++j) {
      v8f z = {0.f, 0.f, 0.f, 0.f, 0.f, 0.f, 0.f, 0.f};
      const bf16_t* krow = &Ks[(j * 16 + ln) * 72];
      z = wmma_bf16(frag_ld(krow, hf), qf0, z);
      z = wmma_bf16(frag_ld(krow + 32, hf), qf1, z);
      sacc[j] = z;
    }

    // Causal mask + online softmax (column q = ln, shared by lanes l,l+16).
    float tmax = mx;
#pragma unroll
    for (int j = 0; j < 4; ++j)
#pragma unroll
      for (int i = 0; i < 8; ++i) {
        const int sg = s0 + j * 16 + i + 8 * hf;
        float val = (sg <= qg) ? sacc[j][i] : -1e30f;
        sacc[j][i] = val;
        tmax = fmaxf(tmax, val);
      }
    tmax = fmaxf(tmax, __shfl_xor(tmax, 16, 32));
    const float corr = __expf(mx - tmax);
    mx = tmax;
    float lsum = 0.0f;
#pragma unroll
    for (int j = 0; j < 4; ++j)
#pragma unroll
      for (int i = 0; i < 8; ++i) {
        const float pe = __expf(sacc[j][i] - mx);
        sacc[j][i] = pe;
        lsum += pe;
      }
    lsum += __shfl_xor(lsum, 16, 32);
    denom = denom * corr + lsum;
#pragma unroll
    for (int j = 0; j < 4; ++j)
#pragma unroll
      for (int i = 0; i < 8; ++i) oacc[j][i] *= corr;

    // Pack P^T into bf16 B-fragments (C layout == B layout, in registers).
    bf16x16 pf0, pf1;
#pragma unroll
    for (int i = 0; i < 8; ++i) {
      pf0[i]     = (bf16_t)sacc[0][i];
      pf0[8 + i] = (bf16_t)sacc[1][i];
      pf1[i]     = (bf16_t)sacc[2][i];
      pf1[8 + i] = (bf16_t)sacc[3][i];
    }

    // O^T += V^T @ P^T, 4 d-subtiles.
#pragma unroll
    for (int jd = 0; jd < 4; ++jd) {
      const bf16_t* vrow = &Vt[(jd * 16 + ln) * 72];
      oacc[jd] = wmma_bf16(frag_ld(vrow, hf), pf0, oacc[jd]);
      oacc[jd] = wmma_bf16(frag_ld(vrow + 32, hf), pf1, oacc[jd]);
    }
  }

  // O^T layout: lane holds (d = jd*16 + i + 8*hf, q = ln).  Write [B,T,H,D].
  const float inv = 1.0f / denom;
  const size_t base = (((size_t)b * TT + qg) * HH + h) * DD;
#pragma unroll
  for (int jd = 0; jd < 4; ++jd) {
    bf16x8 o;
#pragma unroll
    for (int i = 0; i < 8; ++i) o[i] = (bf16_t)(oacc[jd][i] * inv);
    *(bf16x8*)(AO + base + jd * 16 + hf * 8) = o;
  }
}

// ---------------------------------------------------------------------------
// Kernel 3: output projection.  out = AO @ Wo^T + bo, fp32 out.
// Same 128x64 tiling as kernel 1; A is already bf16.
// ---------------------------------------------------------------------------
__global__ __launch_bounds__(256) void mha_outproj_kernel(
    const bf16_t* __restrict__ AO, const float* __restrict__ Wo,
    const float* __restrict__ bo, float* __restrict__ out) {
  __shared__ bf16_t Asm[128 * 40];
  __shared__ bf16_t Bsm[64 * 40];

  const int m0 = blockIdx.x * 128;
  const int n0 = blockIdx.y * 64;
  const int tid = threadIdx.x;
  const int wave = tid >> 5, lane = tid & 31;
  const int wm = wave >> 1, wn = wave & 1;
  const int hf = lane >> 4, ln = lane & 15;

  v8f acc[2][2];
#pragma unroll
  for (int a = 0; a < 2; ++a)
#pragma unroll
    for (int b = 0; b < 2; ++b)
#pragma unroll
      for (int i = 0; i < 8; ++i) acc[a][b][i] = 0.0f;

  for (int kt = 0; kt < CC; kt += 32) {
    __syncthreads();
#pragma unroll
    for (int it = 0; it < 2; ++it) {  // A: 128x32 bf16 = 512 chunks of 8
      int idx = tid + it * 256;
      int r = idx >> 2, c8 = (idx & 3) * 8;
      *(bf16x8*)&Asm[r * 40 + c8] =
          *(const bf16x8*)(AO + (size_t)(m0 + r) * CC + kt + c8);
    }
#pragma unroll
    for (int it = 0; it < 2; ++it) {  // B: 64x32 f32 = 512 float4
      int idx = tid + it * 256;
      int r = idx >> 3, c4 = (idx & 7) * 4;
      const float4 wb = *(const float4*)(Wo + (size_t)(n0 + r) * CC + kt + c4);
      bf16_t* db = &Bsm[r * 40 + c4];
      db[0] = (bf16_t)wb.x; db[1] = (bf16_t)wb.y;
      db[2] = (bf16_t)wb.z; db[3] = (bf16_t)wb.w;
    }
    if (kt + 32 < CC) {
      __builtin_prefetch(Wo + (size_t)(n0 + (tid >> 2)) * CC + kt + 32, 0, 1);
    }
    __syncthreads();
    bf16x16 af[2], bfr[2];
#pragma unroll
    for (int a = 0; a < 2; ++a)
      af[a] = frag_ld(&Asm[(wm * 32 + a * 16 + ln) * 40], hf);
#pragma unroll
    for (int b = 0; b < 2; ++b)
      bfr[b] = frag_ld(&Bsm[(wn * 32 + b * 16 + ln) * 40], hf);
#pragma unroll
    for (int a = 0; a < 2; ++a)
#pragma unroll
      for (int b = 0; b < 2; ++b) acc[a][b] = wmma_bf16(af[a], bfr[b], acc[a][b]);
  }

#pragma unroll
  for (int a = 0; a < 2; ++a)
#pragma unroll
    for (int b = 0; b < 2; ++b) {
      const int n = n0 + wn * 32 + b * 16 + ln;
      const float bn = bo[n];
#pragma unroll
      for (int i = 0; i < 8; ++i) {
        const int m = m0 + wm * 32 + a * 16 + i + 8 * hf;
        out[(size_t)m * CC + n] = acc[a][b][i] + bn;
      }
    }
}

extern "C" void kernel_launch(void* const* d_in, const int* in_sizes, int n_in,
                              void* d_out, int out_size, void* d_ws,
                              size_t ws_size, hipStream_t stream) {
  const float* q  = (const float*)d_in[0];
  const float* k  = (const float*)d_in[1];
  const float* v  = (const float*)d_in[2];
  // d_in[3] = mask (causal tril) -- applied analytically in-kernel.
  const float* Wq = (const float*)d_in[4];
  const float* bq = (const float*)d_in[5];
  const float* Wk = (const float*)d_in[6];
  const float* bk = (const float*)d_in[7];
  const float* Wv = (const float*)d_in[8];
  const float* bv = (const float*)d_in[9];
  const float* Wo = (const float*)d_in[10];
  const float* bo = (const float*)d_in[11];

  const size_t elems = (size_t)BB * TT * CC;  // 4,194,304
  bf16_t* Qh = (bf16_t*)d_ws;
  bf16_t* Kh = Qh + elems;
  bf16_t* Vh = Kh + elems;
  bf16_t* AO = Vh + elems;

  // 1) QKV projections: grid (M/128, N/64, 3)
  mha_proj_kernel<<<dim3(32, 16, 3), 256, 0, stream>>>(
      q, k, v, Wq, bq, Wk, bk, Wv, bv, Qh, Kh, Vh);

  // 2) Flash attention: grid (T/128, B*H)
  mha_attn_kernel<<<dim3(16, 32), 256, 0, stream>>>(Qh, Kh, Vh, AO);

  // 3) Output projection
  mha_outproj_kernel<<<dim3(32, 16), 256, 0, stream>>>(AO, Wo, bo,
                                                       (float*)d_out);
}
